// GraphDiTUnit_50491635532403
// MI455X (gfx1250) — compile-verified, matrix operand order checked by reading, not run on hardware
//
#include <hip/hip_runtime.h>
#include <hip/hip_bf16.h>
#include <math.h>

typedef __attribute__((ext_vector_type(16))) _Float16 v16h;
typedef __attribute__((ext_vector_type(8)))  float    v8f;

#define HDIM   512
#define NHEAD  8
#define HEADD  64
#define EDIM   128
#define BATCH  256
#define SEQ    256
#define NNODE  2
#define EPSLN  1e-5f

// ---------------------------------------------------------------------------
// CDNA5 WMMA lane helpers (16x16x32 f16, wave32)
// A frag: lane L holds row (L&15); 16 f16 elems cover K in pattern:
//   kb = (L>>4)*8 ; elems 0..7 -> K=kb..kb+7 ; elems 8..15 -> K=kb+16..kb+23
// B frag: lane L holds column (L&15) with the same K pattern.
// C/D:    VGPR v -> row v (lanes 0-15) / row v+8 (lanes 16-31); col = lane&15
// ---------------------------------------------------------------------------
__device__ __forceinline__ int kpat(int lane, int t) {
  int kb = (lane >> 4) << 3;
  return (t < 8) ? (kb + t) : (kb + 8 + t); // t>=8 : kb + 16 + (t-8)
}

template <int ACT>
__device__ __forceinline__ float act_apply(float x) {
  if (ACT == 1) return 0.5f * x * (1.0f + erff(x * 0.70710678118f)); // exact gelu
  if (ACT == 2) return 1.0f / (1.0f + __expf(-x));                  // sigmoid
  if (ACT == 3) return tanhf(x);                                    // tanh
  return x;
}

// ---------------------------------------------------------------------------
// Generic GEMM: C[M,O] = act( A[M,K] * W[O,K]^T + bias[O] ) (+ res[M,O])
// Block: 128x64 output tile, 256 threads = 8 waves (4 x 2), each wave a
// 32x32 sub-tile -> 4 WMMAs per k-step with A/B fragment reuse.
// Double-buffered LDS, software pipelined: next tile's global loads are in
// flight while the current tile's WMMAs execute.
// Call-site invariants: M % 128 == 0, K % 32 == 0. Columns guarded for O=8.
// ---------------------------------------------------------------------------
template <int ACT, bool HASRES>
__global__ __launch_bounds__(256)
void gemm_kernel(const float* __restrict__ A, const float* __restrict__ W,
                 const float* __restrict__ bias, const float* __restrict__ res,
                 float* __restrict__ C, int M, int K, int O) {
  __shared__ _Float16 As[2][128][40];
  __shared__ _Float16 Bs[2][64][40];
  const int tid  = threadIdx.x;
  const int lane = tid & 31;
  const int wid  = tid >> 5;
  const int lrow = lane & 15;
  const int half = lane >> 4;
  const int bm = blockIdx.x * 128;
  const int bn = blockIdx.y * 64;
  const int rbase = (wid & 3) * 32;   // wave row tile (32 rows)
  const int nbase = (wid >> 2) * 32;  // wave col tile (32 cols)

  // staging assignments (branchless; W row pre-clamped for the O=8 case;
  // garbage in padded columns only feeds outputs that are never stored)
  const int ldrA = tid >> 1;              // 0..127
  const int ldcA = (tid & 1) * 16;        // 0,16
  const int ldrB = tid >> 2;              // 0..63
  const int ldcB = (tid & 3) * 8;         // 0,8,16,24
  const int gwB  = bn + ldrB;
  const float* srcA = A + (size_t)(bm + ldrA) * K + ldcA;
  const float* srcB = W + (size_t)(gwB < O ? gwB : O - 1) * K + ldcB;

  v8f acc00 = {}; v8f acc01 = {}; v8f acc10 = {}; v8f acc11 = {};

  // preload k-tile 0 into buffer 0
  #pragma unroll
  for (int i = 0; i < 16; ++i) As[0][ldrA][ldcA + i] = (_Float16)srcA[i];
  #pragma unroll
  for (int i = 0; i < 8; ++i)  Bs[0][ldrB][ldcB + i] = (_Float16)srcB[i];
  __syncthreads();

  const int nsteps = K >> 5;
  for (int step = 0; step < nsteps; ++step) {
    const int cur = step & 1;

    // issue next tile's global loads first (latency hidden behind WMMAs)
    float ra[16]; float rb[8];
    const bool more = (step + 1) < nsteps;
    if (more) {
      const int k0 = (step + 1) << 5;
      #pragma unroll
      for (int i = 0; i < 16; ++i) ra[i] = srcA[k0 + i];
      #pragma unroll
      for (int i = 0; i < 8; ++i)  rb[i] = srcB[k0 + i];
    }

    // compute on current buffer
    v16h a0, a1, b0, b1;
    #pragma unroll
    for (int t = 0; t < 16; ++t) {
      int kk = kpat(lane, t);
      a0[t] = As[cur][rbase + lrow][kk];
      a1[t] = As[cur][rbase + 16 + lrow][kk];
      b0[t] = Bs[cur][nbase + lrow][kk];
      b1[t] = Bs[cur][nbase + 16 + lrow][kk];
    }
    acc00 = __builtin_amdgcn_wmma_f32_16x16x32_f16(false, a0, false, b0,
                                                   (short)0, acc00, false, false);
    acc01 = __builtin_amdgcn_wmma_f32_16x16x32_f16(false, a0, false, b1,
                                                   (short)0, acc01, false, false);
    acc10 = __builtin_amdgcn_wmma_f32_16x16x32_f16(false, a1, false, b0,
                                                   (short)0, acc10, false, false);
    acc11 = __builtin_amdgcn_wmma_f32_16x16x32_f16(false, a1, false, b1,
                                                   (short)0, acc11, false, false);

    // convert + store next tile into the alternate buffer
    if (more) {
      #pragma unroll
      for (int i = 0; i < 16; ++i) As[cur ^ 1][ldrA][ldcA + i] = (_Float16)ra[i];
      #pragma unroll
      for (int i = 0; i < 8; ++i)  Bs[cur ^ 1][ldrB][ldcB + i] = (_Float16)rb[i];
    }
    __syncthreads();
  }

  const int gc0 = bn + nbase + lrow;
  const int gc1 = gc0 + 16;
  #pragma unroll
  for (int v = 0; v < 8; ++v) {
    int gr0 = bm + rbase + v + half * 8;   // rows for acc0x
    int gr1 = gr0 + 16;                    // rows for acc1x
    if (gc0 < O) {
      float x = act_apply<ACT>(acc00[v] + bias[gc0]);
      if (HASRES) x += res[(size_t)gr0 * O + gc0];
      C[(size_t)gr0 * O + gc0] = x;
      float y = act_apply<ACT>(acc10[v] + bias[gc0]);
      if (HASRES) y += res[(size_t)gr1 * O + gc0];
      C[(size_t)gr1 * O + gc0] = y;
    }
    if (gc1 < O) {
      float x = act_apply<ACT>(acc01[v] + bias[gc1]);
      if (HASRES) x += res[(size_t)gr0 * O + gc1];
      C[(size_t)gr0 * O + gc1] = x;
      float y = act_apply<ACT>(acc11[v] + bias[gc1]);
      if (HASRES) y += res[(size_t)gr1 * O + gc1];
      C[(size_t)gr1 * O + gc1] = y;
    }
  }
}

// ---------------------------------------------------------------------------
// Self-attention over the action stream (S=256, HD=64), WMMA for QK^T and PV.
// One block = (b, head, 32-query tile). 128 threads = 4 waves.
// Full 32x256 score panel lives in LDS (no online softmax needed).
// qkv layout: [B, S, 3H], Q at col h*64, K at H + h*64, V at 2H + h*64.
// ---------------------------------------------------------------------------
__global__ __launch_bounds__(128)
void attn_kernel(const float* __restrict__ qkv, float* __restrict__ out) {
  __shared__ _Float16 Qs[32][72];
  __shared__ _Float16 KVs[64][72];
  __shared__ float    Sc[32][260];
  const int tid  = threadIdx.x;
  const int lane = tid & 31;
  const int wid  = tid >> 5;
  const int lrow = lane & 15;
  const int half = lane >> 4;
  const int id = blockIdx.x;
  const int qt = id & 7;           // S/32 = 8 query tiles
  const int h  = (id >> 3) & 7;
  const int b  = id >> 6;
  const int q0 = qt * 32;
  const size_t rstride = 3 * HDIM;
  const int rbase = (wid & 1) * 16;
  const int cbase = (wid >> 1) * 32;

  // load Q tile (32x64)
  {
    int r = tid >> 2; int c0 = (tid & 3) * 16;
    const float* src = qkv + (size_t)(b * SEQ + q0 + r) * rstride + h * HEADD + c0;
    #pragma unroll
    for (int i = 0; i < 16; ++i) Qs[r][c0 + i] = (_Float16)src[i];
  }

  // scores: S = (Q K^T) * 1/8
  for (int kt = 0; kt < 4; ++kt) {
    __syncthreads();
    { // load K tile (64x64)
      int r = tid >> 1; int c0 = (tid & 1) * 32;
      const float* src = qkv + (size_t)(b * SEQ + kt * 64 + r) * rstride + HDIM + h * HEADD + c0;
      #pragma unroll
      for (int i = 0; i < 32; ++i) KVs[r][c0 + i] = (_Float16)src[i];
    }
    __syncthreads();
    v8f s0 = {}; v8f s1 = {};
    #pragma unroll
    for (int d0 = 0; d0 < 64; d0 += 32) {
      v16h af, bf0, bf1;
      #pragma unroll
      for (int t = 0; t < 16; ++t) {
        int kk = kpat(lane, t);
        af[t]  = Qs[rbase + lrow][d0 + kk];
        bf0[t] = KVs[cbase + lrow][d0 + kk];
        bf1[t] = KVs[cbase + 16 + lrow][d0 + kk];
      }
      s0 = __builtin_amdgcn_wmma_f32_16x16x32_f16(false, af, false, bf0,
                                                  (short)0, s0, false, false);
      s1 = __builtin_amdgcn_wmma_f32_16x16x32_f16(false, af, false, bf1,
                                                  (short)0, s1, false, false);
    }
    #pragma unroll
    for (int v = 0; v < 8; ++v) {
      int r = rbase + v + half * 8;
      Sc[r][kt * 64 + cbase + lrow]      = s0[v] * 0.125f;
      Sc[r][kt * 64 + cbase + 16 + lrow] = s1[v] * 0.125f;
    }
  }
  __syncthreads();

  // softmax per query row (32 rows, one thread each)
  if (tid < 32) {
    float m = -1e30f;
    for (int k = 0; k < SEQ; ++k) m = fmaxf(m, Sc[tid][k]);
    float s = 0.0f;
    for (int k = 0; k < SEQ; ++k) { float e = __expf(Sc[tid][k] - m); Sc[tid][k] = e; s += e; }
    float inv = 1.0f / s;
    for (int k = 0; k < SEQ; ++k) Sc[tid][k] *= inv;
  }
  __syncthreads();

  // O = P * V
  v8f o0 = {}; v8f o1 = {};
  for (int kt = 0; kt < 4; ++kt) {
    { // load V tile (64 keys x 64 d)
      int r = tid >> 1; int c0 = (tid & 1) * 32;
      const float* src = qkv + (size_t)(b * SEQ + kt * 64 + r) * rstride + 2 * HDIM + h * HEADD + c0;
      #pragma unroll
      for (int i = 0; i < 32; ++i) KVs[r][c0 + i] = (_Float16)src[i];
    }
    __syncthreads();
    #pragma unroll
    for (int k0 = 0; k0 < 64; k0 += 32) {
      v16h af, bf0, bf1;
      #pragma unroll
      for (int t = 0; t < 16; ++t) {
        int kk = kpat(lane, t);
        af[t]  = (_Float16)Sc[rbase + lrow][kt * 64 + k0 + kk];
        bf0[t] = KVs[k0 + kk][cbase + lrow];
        bf1[t] = KVs[k0 + kk][cbase + 16 + lrow];
      }
      o0 = __builtin_amdgcn_wmma_f32_16x16x32_f16(false, af, false, bf0,
                                                  (short)0, o0, false, false);
      o1 = __builtin_amdgcn_wmma_f32_16x16x32_f16(false, af, false, bf1,
                                                  (short)0, o1, false, false);
    }
    __syncthreads();
  }
  #pragma unroll
  for (int v = 0; v < 8; ++v) {
    int qr = q0 + rbase + v + half * 8;
    out[(size_t)(b * SEQ + qr) * HDIM + h * HEADD + cbase + lrow]      = o0[v];
    out[(size_t)(b * SEQ + qr) * HDIM + h * HEADD + cbase + 16 + lrow] = o1[v];
  }
}

// ---------------------------------------------------------------------------
// adaLN: y = (1+scale)*LN(x) + shift ; one block per row of H=512
// cs[b, 0:H]=scale, cs[b, H:2H]=shift
// ---------------------------------------------------------------------------
__global__ __launch_bounds__(256)
void adaln_kernel(const float* __restrict__ x, const float* __restrict__ cs,
                  float* __restrict__ y, int rowsPerB) {
  __shared__ float rs[256], rq[256];
  const int row = blockIdx.x;
  const int b = row / rowsPerB;
  const int tid = threadIdx.x;
  const float* xr = x + (size_t)row * HDIM;
  float v0 = xr[tid], v1 = xr[tid + 256];
  rs[tid] = v0 + v1;
  rq[tid] = v0 * v0 + v1 * v1;
  __syncthreads();
  for (int s = 128; s > 0; s >>= 1) {
    if (tid < s) { rs[tid] += rs[tid + s]; rq[tid] += rq[tid + s]; }
    __syncthreads();
  }
  float mean = rs[0] * (1.0f / HDIM);
  float var  = rq[0] * (1.0f / HDIM) - mean * mean;
  float rstd = rsqrtf(var + EPSLN);
  const float* c = cs + (size_t)b * (2 * HDIM);
  float* yr = y + (size_t)row * HDIM;
  yr[tid]       = (1.0f + c[tid])       * (v0 - mean) * rstd + c[HDIM + tid];
  yr[tid + 256] = (1.0f + c[tid + 256]) * (v1 - mean) * rstd + c[HDIM + tid + 256];
}

__global__ void silu_kernel(const float* __restrict__ x, float* __restrict__ y, int n) {
  int i = blockIdx.x * blockDim.x + threadIdx.x;
  if (i < n) { float v = x[i]; y[i] = v / (1.0f + __expf(-v)); }
}

// ---------------------------------------------------------------------------
// Graph attention core over N=2 nodes. One block per batch b, wave per head.
// ---------------------------------------------------------------------------
__global__ __launch_bounds__(256)
void graph_core_kernel(const float* __restrict__ gq, const float* __restrict__ gk,
                       const float* __restrict__ gv, const float* __restrict__ gate,
                       const float* __restrict__ scl, const float* __restrict__ shf,
                       const float* __restrict__ hs, const float* __restrict__ eb,
                       float* __restrict__ go) {
  __shared__ float Vp[2][HDIM];
  const int b = blockIdx.x;
  const int tid = threadIdx.x;
  for (int idx = tid; idx < 2 * HDIM; idx += 256) {
    int n = idx >> 9; int j = idx & (HDIM - 1);
    float g = gate[(size_t)b * HDIM + j];
    float sc = scl[(size_t)b * HDIM + j];
    float sh = shf[(size_t)b * HDIM + j];
    Vp[n][j] = (gv[(size_t)(b * 2 + n) * HDIM + j] * (1.0f + sc) + sh) * g;
  }
  __syncthreads();
  const int wid = tid >> 5, lane = tid & 31;
  const int head = wid;
  float s00 = 0, s01 = 0, s10 = 0, s11 = 0;
  #pragma unroll
  for (int t = 0; t < 2; ++t) {
    int d = lane * 2 + t;
    float q0 = gq[(size_t)(b * 2 + 0) * HDIM + head * HEADD + d];
    float q1 = gq[(size_t)(b * 2 + 1) * HDIM + head * HEADD + d];
    float k0 = gk[(size_t)(b * 2 + 0) * HDIM + head * HEADD + d];
    float k1 = gk[(size_t)(b * 2 + 1) * HDIM + head * HEADD + d];
    s00 += q0 * k0; s01 += q0 * k1; s10 += q1 * k0; s11 += q1 * k1;
  }
  #pragma unroll
  for (int off = 16; off > 0; off >>= 1) {
    s00 += __shfl_down(s00, off, 32);
    s01 += __shfl_down(s01, off, 32);
    s10 += __shfl_down(s10, off, 32);
    s11 += __shfl_down(s11, off, 32);
  }
  s00 = __shfl(s00, 0, 32); s01 = __shfl(s01, 0, 32);
  s10 = __shfl(s10, 0, 32); s11 = __shfl(s11, 0, 32);
  float f = 0.125f * (1.0f + hs[b * NHEAD + head]);
  float ebv = eb[b * NHEAD + head];
  s00 *= f; s01 = s01 * f + ebv; s10 = s10 * f + ebv; s11 *= f;
  float m0 = fmaxf(s00, s01), e00 = __expf(s00 - m0), e01 = __expf(s01 - m0);
  float i0 = 1.0f / (e00 + e01), a00 = e00 * i0, a01 = e01 * i0;
  float m1 = fmaxf(s10, s11), e10 = __expf(s10 - m1), e11 = __expf(s11 - m1);
  float i1 = 1.0f / (e10 + e11), a10 = e10 * i1, a11 = e11 * i1;
  #pragma unroll
  for (int t = 0; t < 2; ++t) {
    int d = lane * 2 + t;
    int j = head * HEADD + d;
    go[(size_t)(b * 2 + 0) * HDIM + j] = a00 * Vp[0][j] + a01 * Vp[1][j];
    go[(size_t)(b * 2 + 1) * HDIM + j] = a10 * Vp[0][j] + a11 * Vp[1][j];
  }
}

// ---------------------------------------------------------------------------
// Cross-attention core: queries over 2 node keys. One wave per (b, q).
// ---------------------------------------------------------------------------
__global__ __launch_bounds__(256)
void cross_attn_kernel(const float* __restrict__ qc, const float* __restrict__ kc,
                       const float* __restrict__ vc, float* __restrict__ out) {
  const int tid = threadIdx.x, wid = tid >> 5, lane = tid & 31;
  const int i = blockIdx.x * 8 + wid; // < B*S
  const int b = i >> 8;
  const int q = i & 255;
  for (int head = 0; head < NHEAD; ++head) {
    float d0 = 0, d1 = 0;
    #pragma unroll
    for (int t = 0; t < 2; ++t) {
      int d = lane * 2 + t;
      float qq = qc[(size_t)(b * SEQ + q) * HDIM + head * HEADD + d];
      d0 += qq * kc[(size_t)(b * 2 + 0) * HDIM + head * HEADD + d];
      d1 += qq * kc[(size_t)(b * 2 + 1) * HDIM + head * HEADD + d];
    }
    #pragma unroll
    for (int off = 16; off > 0; off >>= 1) {
      d0 += __shfl_down(d0, off, 32);
      d1 += __shfl_down(d1, off, 32);
    }
    d0 = __shfl(d0, 0, 32) * 0.125f;
    d1 = __shfl(d1, 0, 32) * 0.125f;
    float m = fmaxf(d0, d1);
    float e0 = __expf(d0 - m), e1 = __expf(d1 - m);
    float inv = 1.0f / (e0 + e1);
    float a0 = e0 * inv, a1 = e1 * inv;
    #pragma unroll
    for (int t = 0; t < 2; ++t) {
      int d = lane * 2 + t;
      int j = head * HEADD + d;
      out[(size_t)(b * SEQ + q) * HDIM + j] =
          a0 * vc[(size_t)(b * 2 + 0) * HDIM + j] +
          a1 * vc[(size_t)(b * 2 + 1) * HDIM + j];
    }
  }
}

// ---------------------------------------------------------------------------
// Host orchestration
// ---------------------------------------------------------------------------
static inline void launch_gemm(const float* A, const float* W, const float* b,
                               const float* res, float* C, int M, int K, int O,
                               int act, hipStream_t s) {
  dim3 g((M + 127) / 128, (O + 63) / 64);
  dim3 blk(256);
  switch (act) {
    case 1: gemm_kernel<1, false><<<g, blk, 0, s>>>(A, W, b, res, C, M, K, O); break;
    case 2: gemm_kernel<2, false><<<g, blk, 0, s>>>(A, W, b, res, C, M, K, O); break;
    case 3: gemm_kernel<3, false><<<g, blk, 0, s>>>(A, W, b, res, C, M, K, O); break;
    default:
      if (res) gemm_kernel<0, true><<<g, blk, 0, s>>>(A, W, b, res, C, M, K, O);
      else     gemm_kernel<0, false><<<g, blk, 0, s>>>(A, W, b, res, C, M, K, O);
      break;
  }
}

extern "C" void kernel_launch(void* const* d_in, const int* in_sizes, int n_in,
                              void* d_out, int out_size, void* d_ws, size_t ws_size,
                              hipStream_t stream) {
  (void)in_sizes; (void)n_in; (void)out_size; (void)ws_size;
  const float* action = (const float*)d_in[0];
  const float* node   = (const float*)d_in[1];
  const float* edge   = (const float*)d_in[2];
  const float* temb   = (const float*)d_in[3];
  #define P(i) ((const float*)d_in[i])
  float* out = (float*)d_out;

  const int BS = BATCH * SEQ;
  float* ws = (float*)d_ws;
  size_t off = 0;
  auto alloc = [&](size_t n) { float* p = ws + off; off += n; return p; };
  float* siluc = alloc((size_t)BATCH * HDIM);
  float* csb   = alloc((size_t)BATCH * 2 * HDIM);
  float* P0    = alloc((size_t)BS * HDIM);        // x_n / attention output
  float* P1    = alloc((size_t)BS * 3 * HDIM);    // qkv / cross-Q
  float* P2    = alloc((size_t)BS * 4 * HDIM);    // FF hidden
  float* t_nn  = alloc((size_t)BATCH * NNODE * HDIM);
  float* gq    = alloc((size_t)BATCH * NNODE * HDIM);
  float* gk    = alloc((size_t)BATCH * NNODE * HDIM);
  float* gv    = alloc((size_t)BATCH * NNODE * HDIM);
  float* gateb = alloc((size_t)BATCH * HDIM);
  float* sclb  = alloc((size_t)BATCH * HDIM);
  float* shfb  = alloc((size_t)BATCH * HDIM);
  float* hsb   = alloc((size_t)BATCH * NHEAD);
  float* ebb   = alloc((size_t)BATCH * NHEAD);
  float* gob   = alloc((size_t)BATCH * NNODE * HDIM);
  float* ncur  = alloc((size_t)BATCH * NNODE * HDIM);
  float* nffh  = alloc((size_t)BATCH * NNODE * 4 * HDIM);

  // cond = silu(timestep_embed)
  silu_kernel<<<(BATCH * HDIM + 255) / 256, 256, 0, stream>>>(temb, siluc, BATCH * HDIM);

  // ---- action self-attention ----
  launch_gemm(siluc, P(4), P(5), nullptr, csb, BATCH, HDIM, 2 * HDIM, 0, stream);
  adaln_kernel<<<BS, 256, 0, stream>>>(action, csb, P0, SEQ);
  launch_gemm(P0, P(16), P(17), nullptr, P1, BS, HDIM, 3 * HDIM, 0, stream);
  attn_kernel<<<BATCH * NHEAD * (SEQ / 32), 128, 0, stream>>>(P1, P0);
  launch_gemm(P0, P(18), P(19), action, out, BS, HDIM, HDIM, 0, stream);

  // ---- action FF ----
  launch_gemm(siluc, P(6), P(7), nullptr, csb, BATCH, HDIM, 2 * HDIM, 0, stream);
  adaln_kernel<<<BS, 256, 0, stream>>>(out, csb, P0, SEQ);
  launch_gemm(P0, P(24), P(25), nullptr, P2, BS, HDIM, 4 * HDIM, 1, stream);
  launch_gemm(P2, P(26), P(27), out, out, BS, 4 * HDIM, HDIM, 0, stream);

  // ---- node graph attention ----
  launch_gemm(siluc, P(8), P(9), nullptr, csb, BATCH, HDIM, 2 * HDIM, 0, stream);
  adaln_kernel<<<BATCH * NNODE, 256, 0, stream>>>(node, csb, t_nn, NNODE);
  launch_gemm(t_nn, P(36), P(37), nullptr, gq, BATCH * NNODE, HDIM, HDIM, 0, stream);
  launch_gemm(t_nn, P(38), P(39), nullptr, gk, BATCH * NNODE, HDIM, HDIM, 0, stream);
  launch_gemm(t_nn, P(40), P(41), nullptr, gv, BATCH * NNODE, HDIM, HDIM, 0, stream);
  launch_gemm(edge, P(46), P(47), nullptr, gateb, BATCH, EDIM, HDIM, 2, stream); // sigmoid
  launch_gemm(edge, P(48), P(49), nullptr, sclb,  BATCH, EDIM, HDIM, 3, stream); // tanh
  launch_gemm(edge, P(50), P(51), nullptr, shfb,  BATCH, EDIM, HDIM, 0, stream);
  launch_gemm(edge, P(52), P(53), nullptr, hsb,   BATCH, EDIM, NHEAD, 3, stream); // tanh
  launch_gemm(edge, P(44), P(45), nullptr, ebb,   BATCH, EDIM, NHEAD, 0, stream);
  graph_core_kernel<<<BATCH, 256, 0, stream>>>(gq, gk, gv, gateb, sclb, shfb, hsb, ebb, gob);
  launch_gemm(gob, P(42), P(43), node, ncur, BATCH * NNODE, HDIM, HDIM, 0, stream);

  // ---- node FF ----
  launch_gemm(siluc, P(10), P(11), nullptr, csb, BATCH, HDIM, 2 * HDIM, 0, stream);
  adaln_kernel<<<BATCH * NNODE, 256, 0, stream>>>(ncur, csb, t_nn, NNODE);
  launch_gemm(t_nn, P(28), P(29), nullptr, nffh, BATCH * NNODE, HDIM, 4 * HDIM, 1, stream);
  launch_gemm(nffh, P(30), P(31), ncur, ncur, BATCH * NNODE, 4 * HDIM, HDIM, 0, stream);

  // ---- cross attention ----
  launch_gemm(siluc, P(12), P(13), nullptr, csb, BATCH, HDIM, 2 * HDIM, 0, stream);
  adaln_kernel<<<BS, 256, 0, stream>>>(out, csb, P0, SEQ);
  launch_gemm(P0, P(20), P(21), nullptr, P1, BS, HDIM, HDIM, 0, stream); // Q
  const float* Wqkv_c = P(20);
  const float* bqkv_c = P(21);
  launch_gemm(ncur, Wqkv_c + (size_t)HDIM * HDIM,     bqkv_c + HDIM,     nullptr, gk,
              BATCH * NNODE, HDIM, HDIM, 0, stream); // K
  launch_gemm(ncur, Wqkv_c + (size_t)2 * HDIM * HDIM, bqkv_c + 2 * HDIM, nullptr, gv,
              BATCH * NNODE, HDIM, HDIM, 0, stream); // V
  cross_attn_kernel<<<BS / 8, 256, 0, stream>>>(P1, gk, gv, P0);
  launch_gemm(P0, P(22), P(23), out, out, BS, HDIM, HDIM, 0, stream);

  // ---- cross FF ----
  launch_gemm(siluc, P(14), P(15), nullptr, csb, BATCH, HDIM, 2 * HDIM, 0, stream);
  adaln_kernel<<<BS, 256, 0, stream>>>(out, csb, P0, SEQ);
  launch_gemm(P0, P(32), P(33), nullptr, P2, BS, HDIM, 4 * HDIM, 1, stream);
  launch_gemm(P2, P(34), P(35), out, out, BS, 4 * HDIM, HDIM, 0, stream);
  #undef P
}
